// ResidualSSM_56040733278826
// MI455X (gfx1250) — matrix operand order repaired and checked by reading, not compiled, not run on hardware
//
#include <hip/hip_runtime.h>
#include <math.h>

// ---------------------------------------------------------------------------
// MI455X (gfx1250) implementation of the bidirectional selective-SSM model.
// Dense GEMMs: V_WMMA_F32_16X16X32_BF16, 64-row M-tile per block, A slab
// staged through LDS (conflict-free padded layout), B fragments register-
// resident and reused across 4 M-tiles.
// ---------------------------------------------------------------------------

typedef __bf16 bf16;
typedef __attribute__((ext_vector_type(8)))  __bf16 bf16x8;
typedef __attribute__((ext_vector_type(16))) __bf16 bf16x16;
typedef __attribute__((ext_vector_type(8)))  float  f32x8;

constexpr int Bsz   = 64;
constexpr int Tlen  = 512;
constexpr int Dm    = 512;
constexpr int Mrows = Bsz * Tlen;      // 32768 token rows
constexpr int KIN   = 1770;            // 1536 + 234
constexpr int KPAD  = 1792;            // KIN padded to multiple of 128
constexpr int NCLS  = 234;
constexpr int NPAD  = 256;             // classes padded to multiple of 64

// ---------------------------------------------------------------------------
// bf16 WMMA GEMM:  C[M x ldc] = A[M x K] * Bw[N x K]^T (+bias)(+act)
//
// Block tile: 64 rows (MT=4 x 16) x (W * NT * 16) cols, W = waves/block.
// K is chunked by BK=128; the 64 x 128 A slab is cooperatively loaded into
// LDS (rows padded by 16B so the 32-lane ds_load_b128 is conflict-free),
// then each wave streams B fragments straight from global (L2-resident
// weights) into NT register fragments and reuses them across the 4 M-tiles.
// Fragment layouts (CDNA5 16-bit WMMA, wave32, half = lane>>4, l16 = lane&15):
//   A 16x32: lane holds row l16, K = {half*8 .. +7, 16+half*8 .. +7}
//   B 32x16: lane holds weight row l16, K = {half*8 .. +7, 16+half*8 .. +7}
// ---------------------------------------------------------------------------
template <int NT>
__global__ __launch_bounds__(256) void wmma_gemm_bf16_kernel(
    const bf16* __restrict__ A,     // M x K, row-major
    const bf16* __restrict__ Bw,    // N x K, row-major (weight layout)
    const float* __restrict__ bias, // N floats or nullptr
    float* __restrict__ C,          // M x ldc fp32, or nullptr
    bf16* __restrict__ Cbf,         // M x ldc bf16, or nullptr
    int K, int ldc, int n_limit, int act)
{
    constexpr int MT  = 4;          // 16-row M-tiles per block
    constexpr int BK  = 128;        // K chunk
    constexpr int LDA = BK + 8;     // +16B row pad -> conflict-free ds_load_b128
    __shared__ __align__(16) bf16 Asmem[MT * 16 * LDA];

    const int tid  = threadIdx.x;
    const int lane = tid & 31;
    const int wave = tid >> 5;
    const int W    = blockDim.x >> 5;
    const int half = lane >> 4;
    const int l16  = lane & 15;

    const int m0 = blockIdx.x * (MT * 16);
    const int n0 = (blockIdx.y * W + wave) * (16 * NT);

    const bf16* bp0 = Bw + (size_t)(n0 + l16) * K + half * 8;

    f32x8 acc[MT][NT];
#pragma unroll
    for (int mt = 0; mt < MT; ++mt)
#pragma unroll
        for (int i = 0; i < NT; ++i)
            acc[mt][i] = f32x8{0.f,0.f,0.f,0.f,0.f,0.f,0.f,0.f};

    for (int k0 = 0; k0 < K; k0 += BK) {
        // ---- cooperative global -> LDS load of the 64 x BK A slab ----
        for (int e = tid; e < MT * 16 * (BK / 8); e += blockDim.x) {
            int row = e / (BK / 8);
            int c8  = e % (BK / 8);
            const bf16* src = A + (size_t)(m0 + row) * K + k0 + c8 * 8;
            *(bf16x8*)&Asmem[row * LDA + c8 * 8] = *(const bf16x8*)src;
            if (k0 + BK < K) __builtin_prefetch(src + BK, 0, 1);
        }
        __syncthreads();

#pragma unroll
        for (int kk = 0; kk < BK; kk += 32) {
            // B fragments first, into distinct registers (loads clause +
            // overlap; reused across all MT m-tiles)
            bf16x16 bfrag[NT];
#pragma unroll
            for (int i = 0; i < NT; ++i) {
                const bf16* bp = bp0 + (size_t)i * 16 * K + k0 + kk;
                bf16x8 bl = *(const bf16x8*)bp;
                bf16x8 bh = *(const bf16x8*)(bp + 16);
                bfrag[i] = __builtin_shufflevector(bl, bh,
                    0,1,2,3,4,5,6,7,8,9,10,11,12,13,14,15);
            }
#pragma unroll
            for (int mt = 0; mt < MT; ++mt) {
                const bf16* ar = &Asmem[(mt * 16 + l16) * LDA + half * 8 + kk];
                bf16x8 al = *(const bf16x8*)ar;
                bf16x8 ah = *(const bf16x8*)(ar + 16);
                bf16x16 afrag = __builtin_shufflevector(al, ah,
                    0,1,2,3,4,5,6,7,8,9,10,11,12,13,14,15);
#pragma unroll
                for (int i = 0; i < NT; ++i)
                    acc[mt][i] = __builtin_amdgcn_wmma_f32_16x16x32_bf16(
                        false, afrag, false, bfrag[i], (short)0,
                        acc[mt][i], false, false);
            }
        }
        __syncthreads();
    }

    // Epilogue. C/D layout: VGPR r -> row (tile_m + r + half*8), col n + l16
#pragma unroll
    for (int mt = 0; mt < MT; ++mt) {
#pragma unroll
        for (int i = 0; i < NT; ++i) {
            int n = n0 + i * 16 + l16;
            float bv = bias ? bias[n] : 0.f;
#pragma unroll
            for (int r = 0; r < 8; ++r) {
                int m = m0 + mt * 16 + r + half * 8;
                float v = acc[mt][i][r] + bv;
                if (act == 1) v = (v > 20.f) ? v : log1pf(__expf(v)); // softplus
                if (n < n_limit) {
                    size_t o = (size_t)m * ldc + n;
                    if (C)   C[o]   = v;
                    if (Cbf) Cbf[o] = (bf16)v;
                }
            }
        }
    }
}

// ---------------------------------------------------------------------------
// Build bf16 A matrix for the input projection: concat(emb, first_pass), K-pad
// ---------------------------------------------------------------------------
__global__ __launch_bounds__(256) void build_xcat_kernel(
    const float* __restrict__ emb, const float* __restrict__ fp,
    bf16* __restrict__ dst)
{
    size_t i = (size_t)blockIdx.x * blockDim.x + threadIdx.x;
    if (i >= (size_t)Mrows * KPAD) return;
    int c = (int)(i % KPAD);
    size_t r = i / KPAD;
    float v = 0.f;
    if (c < 1536)      v = emb[r * 1536 + c];
    else if (c < KIN)  v = fp[r * 234 + (c - 1536)];
    dst[i] = (bf16)v;
}

// fp32 -> bf16 with row/col padding (weights)
__global__ __launch_bounds__(256) void cvt_pad_kernel(
    const float* __restrict__ src, bf16* __restrict__ dst,
    int src_rows, int src_cols, int dst_rows, int dst_cols)
{
    size_t i = (size_t)blockIdx.x * blockDim.x + threadIdx.x;
    if (i >= (size_t)dst_rows * dst_cols) return;
    int r = (int)(i / dst_cols), c = (int)(i % dst_cols);
    float v = (r < src_rows && c < src_cols) ? src[(size_t)r * src_cols + c] : 0.f;
    dst[i] = (bf16)v;
}

__global__ void pad_bias_kernel(const float* __restrict__ src,
                                float* __restrict__ dst, int n_src, int n_dst)
{
    int i = blockIdx.x * blockDim.x + threadIdx.x;
    if (i < n_dst) dst[i] = (i < n_src) ? src[i] : 0.f;
}

// meta[b, d] = concat(site_emb[site], hour_emb[hour]) @ meta_W[d, :16] + meta_b
__global__ __launch_bounds__(512) void meta_kernel(
    const int* __restrict__ site_ids, const int* __restrict__ hours,
    const float* __restrict__ site_emb, const float* __restrict__ hour_emb,
    const float* __restrict__ mW, const float* __restrict__ mb,
    float* __restrict__ meta)
{
    int b = blockIdx.x, d = threadIdx.x;
    int sid = min(max(site_ids[b], 0), 19);
    int hid = min(max(hours[b], 0), 23);
    float acc = mb[d];
#pragma unroll
    for (int j = 0; j < 8; ++j) acc += mW[d * 16 + j]     * site_emb[sid * 8 + j];
#pragma unroll
    for (int j = 0; j < 8; ++j) acc += mW[d * 16 + 8 + j] * hour_emb[hid * 8 + j];
    meta[b * Dm + d] = acc;
}

__device__ __forceinline__ float wave_reduce_add(float v) {
#pragma unroll
    for (int o = 16; o > 0; o >>= 1) v += __shfl_xor(v, o, 32);
    return v;
}

// LayerNorm over D=512 (one block per row, 8 waves), optional second input,
// optional GELU / pos / meta add; fp32 and/or bf16 outputs.
__global__ __launch_bounds__(256) void ln_kernel(
    const float* __restrict__ X1, const float* __restrict__ X2,
    const float* __restrict__ g, const float* __restrict__ be,
    const float* __restrict__ pos, const float* __restrict__ meta,
    float* __restrict__ Yf, bf16* __restrict__ Ybf, int do_gelu)
{
    __shared__ float red[16];
    int row = blockIdx.x;
    int t = row % Tlen, b = row / Tlen;
    int d0 = threadIdx.x, d1 = threadIdx.x + 256;
    size_t base = (size_t)row * Dm;

    float v0 = X1[base + d0], v1 = X1[base + d1];
    if (X2) { v0 += X2[base + d0]; v1 += X2[base + d1]; }
    float s  = wave_reduce_add(v0 + v1);
    float s2 = wave_reduce_add(v0 * v0 + v1 * v1);
    int w = threadIdx.x >> 5, lane = threadIdx.x & 31;
    if (lane == 0) { red[w] = s; red[8 + w] = s2; }
    __syncthreads();
    if (w == 0) {
        float a  = (lane < 8) ? red[lane] : 0.f;
        float a2 = (lane < 8) ? red[8 + lane] : 0.f;
        a = wave_reduce_add(a); a2 = wave_reduce_add(a2);
        if (lane == 0) { red[0] = a; red[1] = a2; }
    }
    __syncthreads();
    float mean = red[0] * (1.f / Dm);
    float var  = red[1] * (1.f / Dm) - mean * mean;
    float rstd = rsqrtf(var + 1e-5f);

#pragma unroll
    for (int h = 0; h < 2; ++h) {
        int d = h ? d1 : d0;
        float x = h ? v1 : v0;
        float y = (x - mean) * rstd * g[d] + be[d];
        if (do_gelu) y = 0.5f * y * (1.f + erff(y * 0.70710678118f));
        if (pos)  y += pos[(size_t)t * Dm + d];
        if (meta) y += meta[(size_t)b * Dm + d];
        if (Yf)  Yf[base + d]  = y;
        if (Ybf) Ybf[base + d] = (bf16)y;
    }
}

// Depthwise causal conv (k=4) + bias + SiLU.  dir=0: causal window (fwd),
// dir=1: anti-causal window (equivalent to conv on the time-reversed seq).
__global__ __launch_bounds__(256) void conv_silu_kernel(
    const float* __restrict__ Xs, const float* __restrict__ convW,
    const float* __restrict__ convb, float* __restrict__ Xc,
    bf16* __restrict__ Xc_bf, int dir)
{
    size_t idx = (size_t)blockIdx.x * blockDim.x + threadIdx.x;
    if (idx >= (size_t)Mrows * Dm) return;
    int d = (int)(idx & (Dm - 1));
    int r = (int)(idx >> 9);
    int t = r % Tlen;
    float acc = convb[d];
#pragma unroll
    for (int j = 0; j < 4; ++j) {
        int tt = dir ? (t + 3 - j) : (t - 3 + j);
        if (tt >= 0 && tt < Tlen)
            acc += convW[d * 4 + j] * Xs[(size_t)(r - t + tt) * Dm + d];
    }
    float sv = acc / (1.f + __expf(-acc));   // SiLU
    Xc[idx] = sv;
    Xc_bf[idx] = (bf16)sv;
}

// Selective scan.  One thread per (b, d) pair; 16 SSM states in registers.
// dt already has softplus applied (GEMM epilogue).  Writes y + x*D directly
// into the bf16 concat buffer at column offset dir*512, at ORIGINAL t index.
__global__ __launch_bounds__(256) void ssm_scan_kernel(
    const float* __restrict__ Xin,   // SSM input h (Hres), M x 512
    const float* __restrict__ dt,    // softplus(dt), M x 512
    const float* __restrict__ BC,    // M x 32 : [B_t(16) | C_t(16)]
    const float* __restrict__ A_log, // 512 x 16
    const float* __restrict__ Dp,    // 512
    bf16* __restrict__ Ycat,         // M x 1024 bf16
    int dir)
{
    int idx = blockIdx.x * blockDim.x + threadIdx.x;   // 0 .. 32767
    int d = idx & (Dm - 1);
    int b = idx >> 9;

    float A[16], st[16];
#pragma unroll
    for (int n = 0; n < 16; ++n) {
        A[n]  = -__expf(A_log[d * 16 + n]);
        st[n] = 0.f;
    }
    float Dd = Dp[d];

    for (int step = 0; step < Tlen; ++step) {
        int t = dir ? (Tlen - 1 - step) : step;
        size_t r = (size_t)b * Tlen + t;
        float dtv = dt[r * Dm + d];
        float xv  = Xin[r * Dm + d];
        const float* bc = BC + r * 32;
        float xdt = xv * dtv;
        float y = 0.f;
#pragma unroll
        for (int n = 0; n < 16; ++n) {
            float dA = __expf(A[n] * dtv);
            st[n] = st[n] * dA + xdt * bc[n];
            y += st[n] * bc[16 + n];
        }
        Ycat[r * 1024 + dir * Dm + d] = (bf16)(y + xv * Dd);
    }
}

// ---------------------------------------------------------------------------
extern "C" void kernel_launch(void* const* d_in, const int* in_sizes, int n_in,
                              void* d_out, int out_size, void* d_ws, size_t ws_size,
                              hipStream_t stream)
{
    (void)in_sizes; (void)n_in; (void)out_size; (void)ws_size;

    // Flattened input order (setup_inputs dict order, params nested in order)
    const float* emb      = (const float*)d_in[0];
    const float* fpass    = (const float*)d_in[1];
    const int*   site_ids = (const int*)  d_in[2];
    const int*   hours    = (const int*)  d_in[3];
    const float* inp_W    = (const float*)d_in[4];
    const float* inp_b    = (const float*)d_in[5];
    const float* inp_g    = (const float*)d_in[6];
    const float* inp_be   = (const float*)d_in[7];
    const float* site_emb = (const float*)d_in[8];
    const float* hour_emb = (const float*)d_in[9];
    const float* meta_W   = (const float*)d_in[10];
    const float* meta_b   = (const float*)d_in[11];
    const float* pos_enc  = (const float*)d_in[12];
    // ssm_fwd: indices 13..21, ssm_bwd: 22..30
    const float* merge_W  = (const float*)d_in[31];
    const float* merge_b  = (const float*)d_in[32];
    const float* norm_g   = (const float*)d_in[33];
    const float* norm_b   = (const float*)d_in[34];
    const float* out_W    = (const float*)d_in[35];
    const float* out_b    = (const float*)d_in[36];
    float* out = (float*)d_out;

    // ---- workspace carve (~630 MB total) ----
    char* ws = (char*)d_ws;
    size_t off = 0;
    auto carve = [&](size_t bytes) -> char* {
        char* p = ws + off;
        off += (bytes + 255) & ~(size_t)255;
        return p;
    };
    bf16*  Xcat    = (bf16*) carve((size_t)Mrows * KPAD * 2);
    bf16*  Win_bf  = (bf16*) carve((size_t)Dm * KPAD * 2);
    float* Hm      = (float*)carve((size_t)Mrows * Dm * 4);   // pre-LN h / merge out
    float* Hres    = (float*)carve((size_t)Mrows * Dm * 4);
    bf16*  Hres_bf = (bf16*) carve((size_t)Mrows * Dm * 2);
    float* metaB   = (float*)carve((size_t)Bsz * Dm * 4);
    float* Xs      = (float*)carve((size_t)Mrows * Dm * 4);
    float* Xc      = (float*)carve((size_t)Mrows * Dm * 4);
    bf16*  Xc_bf   = (bf16*) carve((size_t)Mrows * Dm * 2);
    float* dtb     = (float*)carve((size_t)Mrows * Dm * 4);
    float* BC      = (float*)carve((size_t)Mrows * 32 * 4);
    bf16*  Ycat    = (bf16*) carve((size_t)Mrows * 1024 * 2);
    bf16*  Hn_bf   = (bf16*) carve((size_t)Mrows * Dm * 2);
    bf16*  Wssm_bf = (bf16*) carve((size_t)Dm * Dm * 2);
    bf16*  dtW_bf  = (bf16*) carve((size_t)Dm * Dm * 2);
    bf16*  BCw_bf  = (bf16*) carve((size_t)32 * Dm * 2);
    bf16*  mrgW_bf = (bf16*) carve((size_t)Dm * 1024 * 2);
    bf16*  outW_bf = (bf16*) carve((size_t)NPAD * Dm * 2);
    float* outb_p  = (float*)carve((size_t)NPAD * 4);

    const dim3 blk256(256);
    auto grid1d = [](size_t n) { return dim3((unsigned)((n + 255) / 256)); };
    const dim3 ggrid(Mrows / 64, 1);   // 64-row M-tile per block

    // ---- Stage 0: build bf16 operands ----
    build_xcat_kernel<<<grid1d((size_t)Mrows * KPAD), blk256, 0, stream>>>(emb, fpass, Xcat);
    cvt_pad_kernel<<<grid1d((size_t)Dm * KPAD), blk256, 0, stream>>>(
        inp_W, Win_bf, Dm, KIN, Dm, KPAD);

    // ---- Stage 1: input projection (M x 1792) x (512 x 1792)^T -> Hm ----
    wmma_gemm_bf16_kernel<4><<<ggrid, 256, 0, stream>>>(
        Xcat, Win_bf, inp_b, Hm, nullptr, KPAD, Dm, Dm, 0);

    meta_kernel<<<dim3(Bsz), 512, 0, stream>>>(
        site_ids, hours, site_emb, hour_emb, meta_W, meta_b, metaB);

    // LN + GELU + pos + meta -> Hres (fp32) + Hres_bf
    ln_kernel<<<dim3(Mrows), blk256, 0, stream>>>(
        Hm, nullptr, inp_g, inp_be, pos_enc, metaB, Hres, Hres_bf, 1);

    // ---- Stage 2: bidirectional selective SSM ----
    for (int dir = 0; dir < 2; ++dir) {
        int pb = 13 + dir * 9;
        const float* s_inW   = (const float*)d_in[pb + 0]; // (1024,512); rows 0..511 used
        const float* s_convW = (const float*)d_in[pb + 1]; // (512,1,4)
        const float* s_convb = (const float*)d_in[pb + 2];
        const float* s_dtW   = (const float*)d_in[pb + 3]; // (512,512)
        const float* s_dtb   = (const float*)d_in[pb + 4];
        const float* s_Alog  = (const float*)d_in[pb + 5]; // (512,16)
        const float* s_D     = (const float*)d_in[pb + 6];
        const float* s_BW    = (const float*)d_in[pb + 7]; // (16,512)
        const float* s_CW    = (const float*)d_in[pb + 8]; // (16,512)

        cvt_pad_kernel<<<grid1d((size_t)Dm * Dm), blk256, 0, stream>>>(
            s_inW, Wssm_bf, Dm, Dm, Dm, Dm);                 // first 512 rows only
        cvt_pad_kernel<<<grid1d((size_t)Dm * Dm), blk256, 0, stream>>>(
            s_dtW, dtW_bf, Dm, Dm, Dm, Dm);
        cvt_pad_kernel<<<grid1d((size_t)16 * Dm), blk256, 0, stream>>>(
            s_BW, BCw_bf, 16, Dm, 16, Dm);
        cvt_pad_kernel<<<grid1d((size_t)16 * Dm), blk256, 0, stream>>>(
            s_CW, BCw_bf + (size_t)16 * Dm, 16, Dm, 16, Dm);

        // x_ssm = Hres @ in_W[:512,:]^T
        wmma_gemm_bf16_kernel<4><<<ggrid, 256, 0, stream>>>(
            Hres_bf, Wssm_bf, nullptr, Xs, nullptr, Dm, Dm, Dm, 0);

        conv_silu_kernel<<<grid1d((size_t)Mrows * Dm), blk256, 0, stream>>>(
            Xs, s_convW, s_convb, Xc, Xc_bf, dir);

        // dt = softplus(xc @ dt_W^T + dt_b)  (softplus fused in epilogue)
        wmma_gemm_bf16_kernel<4><<<ggrid, 256, 0, stream>>>(
            Xc_bf, dtW_bf, s_dtb, dtb, nullptr, Dm, Dm, Dm, 1);

        // [B_t | C_t] = xc @ [B_W;C_W]^T   (N=32, 2 waves x 1 tile each)
        wmma_gemm_bf16_kernel<1><<<ggrid, 64, 0, stream>>>(
            Xc_bf, BCw_bf, nullptr, BC, nullptr, Dm, 32, 32, 0);

        ssm_scan_kernel<<<dim3(Mrows * Dm / (Tlen * 256)), blk256, 0, stream>>>(
            Hres, dtb, BC, s_Alog, s_D, Ycat, dir);
    }

    // ---- Stage 3: merge + residual LN ----
    cvt_pad_kernel<<<grid1d((size_t)Dm * 1024), blk256, 0, stream>>>(
        merge_W, mrgW_bf, Dm, 1024, Dm, 1024);
    wmma_gemm_bf16_kernel<4><<<ggrid, 256, 0, stream>>>(
        Ycat, mrgW_bf, merge_b, Hm, nullptr, 1024, Dm, Dm, 0);
    ln_kernel<<<dim3(Mrows), blk256, 0, stream>>>(
        Hm, Hres, norm_g, norm_b, nullptr, nullptr, nullptr, Hn_bf, 0);

    // ---- Stage 4: classifier head (N padded 234 -> 256, ragged stores) ----
    cvt_pad_kernel<<<grid1d((size_t)NPAD * Dm), blk256, 0, stream>>>(
        out_W, outW_bf, NCLS, Dm, NPAD, Dm);
    pad_bias_kernel<<<grid1d(NPAD), blk256, 0, stream>>>(out_b, outb_p, NCLS, NPAD);
    wmma_gemm_bf16_kernel<4><<<ggrid, 128, 0, stream>>>(
        Hn_bf, outW_bf, outb_p, out, nullptr, Dm, NCLS, NCLS, 0);
}
